// PartitionAttention1d_68624987456222
// MI455X (gfx1250) — matrix operand order repaired and checked by reading, not compiled, not run on hardware
//
#include <hip/hip_runtime.h>
#include <hip/hip_bf16.h>
#include <math.h>

// ---------------------------------------------------------------------------
// Types for CDNA5 WMMA (wave32, 16x16x32 bf16 -> f32)
// ---------------------------------------------------------------------------
typedef __attribute__((ext_vector_type(16))) __bf16 v16bf;
typedef __attribute__((ext_vector_type(8)))  __bf16 v8bf;
typedef __attribute__((ext_vector_type(8)))  float  v8f;
typedef int v4i __attribute__((vector_size(16)));

union V16 { v16bf v; v8bf h[2]; };

__device__ __forceinline__ v8f wmma_bf16(v16bf a, v16bf b, v8f c) {
  // D = A(16x32) * B(32x16) + C(16x16), fp32 accumulate
  return __builtin_amdgcn_wmma_f32_16x16x32_bf16(
      /*neg_a=*/false, a, /*neg_b=*/false, b,
      /*c_mod=*/(short)0, c, /*reuse_a=*/false, /*reuse_b=*/false);
}

// A-fragment (16x32 bf16) from a row-major bf16 matrix, ld = K.
// Lane layout (ISA 7.12.2): row = lane&15; elems 0..7 = K kbase..kbase+7,
// elems 8..15 = K kbase+16..kbase+23, kbase = (lane&16)?8:0.
__device__ __forceinline__ v16bf load_afrag(const __bf16* __restrict__ A,
                                            int row, int ldk, int kt, int lane) {
  int kbase = kt * 32 + ((lane & 16) ? 8 : 0);
  const __bf16* p = A + (size_t)row * ldk + kbase;
  V16 u;
  u.h[0] = *reinterpret_cast<const v8bf*>(p);
  u.h[1] = *reinterpret_cast<const v8bf*>(p + 16);
  return u.v;
}

// ---------------------------------------------------------------------------
// Async global->LDS staging (CDNA5 ASYNCcnt path) with safe fallback
// ---------------------------------------------------------------------------
__device__ __forceinline__ void stage16(const __bf16* g, __bf16* l) {
#if __has_builtin(__builtin_amdgcn_global_load_async_to_lds_b128)
  __builtin_amdgcn_global_load_async_to_lds_b128(
      (__attribute__((address_space(1))) v4i*)g,
      (__attribute__((address_space(3))) v4i*)l, 0, 0);
#else
  *reinterpret_cast<v8bf*>(l) = *reinterpret_cast<const v8bf*>(g);
#endif
}

__device__ __forceinline__ void stage_wait() {
#if __has_builtin(__builtin_amdgcn_global_load_async_to_lds_b128)
#if __has_builtin(__builtin_amdgcn_s_wait_asynccnt)
  __builtin_amdgcn_s_wait_asynccnt(0);
#else
  asm volatile("s_wait_asynccnt 0" ::: "memory");
#endif
#endif
}

// ---------------------------------------------------------------------------
// Problem constants
// ---------------------------------------------------------------------------
#define NB    16
#define LSEQ  8192
#define CDIM  256
#define ROWS  131072      // NB*LSEQ
#define WS64  64          // window size
#define NWIN  2048        // NB * (LSEQ/WS64)
#define NHEAD 8
#define DHEAD 32
#define HDIM  1024
#define KCH   256         // GEMM K staged per LDS chunk (8 k-tiles)

// ---------------------------------------------------------------------------
// Weight repack: row-major fp32 W[K][N] -> bf16 B-fragment tiles.
// Packed index: (((ntile*(K/32)+kt)*32 + lane)*16 + e)
//   k = kt*32 + (e&7) + ((lane&16)?8:0) + ((e&8)?16:0),  n = ntile*16 + (lane&15)
// ---------------------------------------------------------------------------
__global__ __launch_bounds__(256) void pack_w_kernel(
    const float* __restrict__ W, __bf16* __restrict__ out, int K, int N) {
  int idx = blockIdx.x * 256 + threadIdx.x;
  int total = K * N;
  if (idx >= total) return;
  int e  = idx & 15;
  int l  = (idx >> 4) & 31;
  int t  = idx >> 9;
  int ktn = K >> 5;
  int kt = t % ktn;
  int nt = t / ktn;
  int k = kt * 32 + (e & 7) + ((l & 16) ? 8 : 0) + ((e & 8) ? 16 : 0);
  int n = nt * 16 + (l & 15);
  out[idx] = (__bf16)W[(size_t)k * N + n];
}

// ---------------------------------------------------------------------------
// LayerNorm over C=256: one wave per row -> bf16 output
// ---------------------------------------------------------------------------
__global__ __launch_bounds__(256) void ln_kernel(
    const float* __restrict__ in, const float* __restrict__ g,
    const float* __restrict__ bvec, __bf16* __restrict__ out, int rows) {
  int lane = threadIdx.x & 31;
  int wv   = threadIdx.x >> 5;
  int row  = blockIdx.x * 8 + wv;
  if (row >= rows) return;
  const float* p = in + (size_t)row * CDIM + lane * 8;
  float4 a = *reinterpret_cast<const float4*>(p);
  float4 b = *reinterpret_cast<const float4*>(p + 4);
  float x[8] = {a.x, a.y, a.z, a.w, b.x, b.y, b.z, b.w};
  float s = 0.f, sq = 0.f;
#pragma unroll
  for (int i = 0; i < 8; ++i) { s += x[i]; sq += x[i] * x[i]; }
#pragma unroll
  for (int m = 16; m >= 1; m >>= 1) {
    s  += __shfl_xor(s,  m, 32);
    sq += __shfl_xor(sq, m, 32);
  }
  float mean = s * (1.0f / CDIM);
  float var  = sq * (1.0f / CDIM) - mean * mean;
  float rstd = rsqrtf(var + 1e-5f);
#pragma unroll
  for (int i = 0; i < 8; ++i) {
    int c = lane * 8 + i;
    out[(size_t)row * CDIM + c] = (__bf16)((x[i] - mean) * rstd * g[c] + bvec[c]);
  }
}

// ---------------------------------------------------------------------------
// WMMA GEMM: C[M,N] = A[M,K](bf16 row-major) @ Bpacked + bias, + epilogue.
// Block: 8 waves; block tile 256(M) x 64(N); wave owns 32 rows (2 M-tiles).
// B strip for the 64 columns is staged into LDS in 256-wide K chunks via
// async global->LDS copies; all waves read B fragments from LDS.
// EPI 0: out_f32 = val + res_f32   (proj / mlp2 residual path)
// EPI 1: out_bf16 = gelu_exact(val)
// EPI 2: QKV scatter into Q/K row-major + V transposed, per (window, head)
// ---------------------------------------------------------------------------
template <int EPI>
__global__ __launch_bounds__(256) void gemm_kernel(
    const __bf16* __restrict__ A, const __bf16* __restrict__ Bp,
    const float* __restrict__ bias, int M, int N, int K,
    const float* __restrict__ res, float* __restrict__ outf,
    __bf16* __restrict__ outb,
    __bf16* __restrict__ qb, __bf16* __restrict__ kb, __bf16* __restrict__ vt) {
  // 4 ntiles x 8 ktiles x 512 elems = 32 KB
  __shared__ alignas(16) __bf16 Bs[4 * (KCH / 32) * 512];

  int lane = threadIdx.x & 31;
  int wave = threadIdx.x >> 5;
  int nstrips = N >> 6;
  int nblk = blockIdx.x % nstrips;
  int mblk = blockIdx.x / nstrips;
  int row0 = mblk * 256 + wave * 32;
  int col0 = nblk * 64;
  int ktn  = K >> 5;

  v8f acc[2][4];
#pragma unroll
  for (int mt = 0; mt < 2; ++mt)
#pragma unroll
    for (int j = 0; j < 4; ++j) acc[mt][j] = (v8f){};

  int ar0 = row0 + (lane & 15);
  for (int k0 = 0; k0 < K; k0 += KCH) {
    if (k0 > 0) __syncthreads();  // previous chunk fully consumed
    // Stage 4 ntile segments (each 8 k-tiles * 512 elems, contiguous) to LDS
    for (int c = threadIdx.x; c < 2048; c += 256) {
      int j  = c >> 9;          // ntile within strip
      int cc = c & 511;         // 16B chunk within segment
      const __bf16* src =
          Bp + ((size_t)(nblk * 4 + j) * ktn + (k0 >> 5)) * 512 + cc * 8;
      stage16(src, Bs + j * 4096 + cc * 8);
    }
    stage_wait();
    __syncthreads();

#pragma unroll
    for (int ktl = 0; ktl < KCH / 32; ++ktl) {
      int kt = (k0 >> 5) + ktl;
      v16bf a0 = load_afrag(A, ar0, K, kt, lane);
      v16bf a1 = load_afrag(A, ar0 + 16, K, kt, lane);
#pragma unroll
      for (int j = 0; j < 4; ++j) {
        const __bf16* p = Bs + ((j * (KCH / 32) + ktl) * 512 + lane * 16);
        V16 u;
        u.h[0] = *reinterpret_cast<const v8bf*>(p);
        u.h[1] = *reinterpret_cast<const v8bf*>(p + 8);
        acc[0][j] = wmma_bf16(a0, u.v, acc[0][j]);
        acc[1][j] = wmma_bf16(a1, u.v, acc[1][j]);
      }
    }
  }

#pragma unroll
  for (int mt = 0; mt < 2; ++mt) {
    int rbase = row0 + mt * 16 + ((lane & 16) ? 8 : 0);
#pragma unroll
    for (int j = 0; j < 4; ++j) {
      int col = col0 + j * 16 + (lane & 15);
      float bv = bias[col];
#pragma unroll
      for (int v = 0; v < 8; ++v) {
        int r = rbase + v;
        float val = acc[mt][j][v] + bv;
        if (EPI == 0) {
          outf[(size_t)r * N + col] = val + res[(size_t)r * N + col];
        } else if (EPI == 1) {
          float gl = 0.5f * val * (1.0f + erff(val * 0.70710678118654752f));
          outb[(size_t)r * N + col] = (__bf16)gl;
        } else {
          // col = head*96 + sel*32 + d  (reshape (nh, 3, DIM_HEAD))
          int hh  = col / 96;
          int rem = col - hh * 96;
          int sel = rem >> 5;
          int d   = rem & 31;
          int win = r >> 6;
          int pos = r & 63;
          __bf16 bw = (__bf16)val;
          size_t qidx = (((size_t)win * NHEAD + hh) * WS64 + pos) * DHEAD + d;
          if (sel == 0)      qb[qidx] = bw;
          else if (sel == 1) kb[qidx] = bw;
          else               vt[(((size_t)win * NHEAD + hh) * DHEAD + d) * WS64 + pos] = bw;
        }
      }
    }
  }
}

// ---------------------------------------------------------------------------
// Windowed attention: one block per window, one wave per head.
// S = Q Kt * scale (WMMA), fp32 softmax via lane shuffles, P (bf16, LDS) @ V
// (V pre-transposed) via WMMA. Output row-major (pos, C) bf16 for proj GEMM.
// ---------------------------------------------------------------------------
__global__ __launch_bounds__(256) void attn_kernel(
    const __bf16* __restrict__ qb, const __bf16* __restrict__ kb,
    const __bf16* __restrict__ vt, __bf16* __restrict__ attnout) {
  __shared__ alignas(16) __bf16 Plds[NHEAD][16][WS64];
  int lane = threadIdx.x & 31;
  int hh   = threadIdx.x >> 5;
  int win  = blockIdx.x;
  int lhalf = (lane & 16) ? 1 : 0;
  const __bf16* Q  = qb + ((size_t)win * NHEAD + hh) * WS64 * DHEAD;
  const __bf16* Km = kb + ((size_t)win * NHEAD + hh) * WS64 * DHEAD;
  const __bf16* Vt = vt + ((size_t)win * NHEAD + hh) * DHEAD * WS64;
  __bf16* Ob = attnout + (size_t)win * WS64 * CDIM + hh * DHEAD;
  const float scale = 0.17677669529663687f;  // 1/sqrt(32)

  for (int qt = 0; qt < 4; ++qt) {
    // A fragment of Q (16 queries x 32 dims); single K-step (DHEAD==32)
    int qrow = qt * 16 + (lane & 15);
    V16 ua;
    int kbq = lhalf ? 8 : 0;
    ua.h[0] = *reinterpret_cast<const v8bf*>(Q + qrow * DHEAD + kbq);
    ua.h[1] = *reinterpret_cast<const v8bf*>(Q + qrow * DHEAD + kbq + 16);

    v8f sacc[4];
#pragma unroll
    for (int kt2 = 0; kt2 < 4; ++kt2) {
      // B fragment of K^T: lane = key column, elems = dims dof..dof+7, dof+16..dof+23 of K row
      int key = kt2 * 16 + (lane & 15);
      int dof = lhalf ? 8 : 0;
      V16 ub;
      ub.h[0] = *reinterpret_cast<const v8bf*>(Km + key * DHEAD + dof);
      ub.h[1] = *reinterpret_cast<const v8bf*>(Km + key * DHEAD + dof + 16);
      v8f z = (v8f){};
      sacc[kt2] = wmma_bf16(ua.v, ub.v, z);
    }

    // Softmax over 64 keys; row q lives in a fixed 16-lane half.
    int rrow = lhalf ? 8 : 0;
#pragma unroll
    for (int v = 0; v < 8; ++v) {
      float x0 = sacc[0][v] * scale, x1 = sacc[1][v] * scale;
      float x2 = sacc[2][v] * scale, x3 = sacc[3][v] * scale;
      float m = fmaxf(fmaxf(x0, x1), fmaxf(x2, x3));
#pragma unroll
      for (int msk = 8; msk >= 1; msk >>= 1) m = fmaxf(m, __shfl_xor(m, msk, 32));
      float e0 = __expf(x0 - m), e1 = __expf(x1 - m);
      float e2 = __expf(x2 - m), e3 = __expf(x3 - m);
      float sum = e0 + e1 + e2 + e3;
#pragma unroll
      for (int msk = 8; msk >= 1; msk >>= 1) sum += __shfl_xor(sum, msk, 32);
      float inv = 1.0f / sum;
      int cr = rrow + v, cc = lane & 15;
      Plds[hh][cr][0 * 16 + cc] = (__bf16)(e0 * inv);
      Plds[hh][cr][1 * 16 + cc] = (__bf16)(e1 * inv);
      Plds[hh][cr][2 * 16 + cc] = (__bf16)(e2 * inv);
      Plds[hh][cr][3 * 16 + cc] = (__bf16)(e3 * inv);
    }

    // O(16x32) = P(16x64) @ V(64x32); V accessed via transposed layout
    v8f oacc[2];
    oacc[0] = (v8f){}; oacc[1] = (v8f){};
#pragma unroll
    for (int kt2 = 0; kt2 < 2; ++kt2) {
      const __bf16* prow = &Plds[hh][lane & 15][0];
      int pb = kt2 * 32 + (lhalf ? 8 : 0);
      V16 up;
      up.h[0] = *reinterpret_cast<const v8bf*>(prow + pb);
      up.h[1] = *reinterpret_cast<const v8bf*>(prow + pb + 16);
#pragma unroll
      for (int nt = 0; nt < 2; ++nt) {
        int kb2 = kt2 * 32 + (lhalf ? 8 : 0);
        const __bf16* vp = Vt + (size_t)(nt * 16 + (lane & 15)) * WS64 + kb2;
        V16 uv;
        uv.h[0] = *reinterpret_cast<const v8bf*>(vp);
        uv.h[1] = *reinterpret_cast<const v8bf*>(vp + 16);
        oacc[nt] = wmma_bf16(up.v, uv.v, oacc[nt]);
      }
    }

#pragma unroll
    for (int nt = 0; nt < 2; ++nt)
#pragma unroll
      for (int v = 0; v < 8; ++v) {
        int q = qt * 16 + rrow + v;
        int d = nt * 16 + (lane & 15);
        Ob[(size_t)q * CDIM + d] = (__bf16)oacc[nt][v];
      }
  }
}

// ---------------------------------------------------------------------------
// Host-side launch
// ---------------------------------------------------------------------------
extern "C" void kernel_launch(void* const* d_in, const int* in_sizes, int n_in,
                              void* d_out, int out_size, void* d_ws, size_t ws_size,
                              hipStream_t stream) {
  (void)in_sizes; (void)n_in; (void)out_size; (void)ws_size;
  const float* x      = (const float*)d_in[0];
  const float* ln1_g  = (const float*)d_in[1];
  const float* ln1_b  = (const float*)d_in[2];
  const float* qkv_w  = (const float*)d_in[3];
  const float* qkv_b  = (const float*)d_in[4];
  const float* proj_w = (const float*)d_in[5];
  const float* proj_b = (const float*)d_in[6];
  const float* ln2_g  = (const float*)d_in[7];
  const float* ln2_b  = (const float*)d_in[8];
  const float* w1     = (const float*)d_in[9];
  const float* b1     = (const float*)d_in[10];
  const float* w2     = (const float*)d_in[11];
  const float* b2     = (const float*)d_in[12];
  float* out = (float*)d_out;

  // Workspace partition (bytes)
  char* ws = (char*)d_ws;
  const size_t SZ_A1 = (size_t)ROWS * HDIM * 2;   // 256 MB
  const size_t SZ_C  = (size_t)ROWS * CDIM * 2;   //  64 MB
  __bf16* a1   = (__bf16*)(ws);
  __bf16* qb   = (__bf16*)(ws + SZ_A1);
  __bf16* kbuf = (__bf16*)(ws + SZ_A1 + SZ_C);
  __bf16* vtb  = (__bf16*)(ws + SZ_A1 + 2 * SZ_C);
  __bf16* hbuf = (__bf16*)(ws + SZ_A1 + 3 * SZ_C);   // h (LN1) then attnout
  char*   wsw  = ws + SZ_A1 + 4 * SZ_C;
  __bf16* wqp  = (__bf16*)(wsw);                      // 256x768
  __bf16* wpp  = (__bf16*)(wsw + 256 * 768 * 2);      // 256x256
  __bf16* w1p  = (__bf16*)(wsw + 256 * 768 * 2 + 256 * 256 * 2);
  __bf16* w2p  = (__bf16*)(wsw + 256 * 768 * 2 + 256 * 256 * 2 + 256 * 1024 * 2);
  __bf16* h2   = qb;   // reuse Q region after attention consumed it

  // 1) Pack weights to WMMA B-fragment layout (bf16)
  pack_w_kernel<<<(256 * 768 + 255) / 256, 256, 0, stream>>>(qkv_w, wqp, 256, 768);
  pack_w_kernel<<<(256 * 256 + 255) / 256, 256, 0, stream>>>(proj_w, wpp, 256, 256);
  pack_w_kernel<<<(256 * 1024 + 255) / 256, 256, 0, stream>>>(w1, w1p, 256, 1024);
  pack_w_kernel<<<(1024 * 256 + 255) / 256, 256, 0, stream>>>(w2, w2p, 1024, 256);

  // 2) LN1 -> bf16
  ln_kernel<<<ROWS / 8, 256, 0, stream>>>(x, ln1_g, ln1_b, hbuf, ROWS);

  // 3) QKV GEMM (131072x256 @ 256x768), scatter to Q/K row-major + V transposed
  gemm_kernel<2><<<(ROWS / 256) * (768 / 64), 256, 0, stream>>>(
      hbuf, wqp, qkv_b, ROWS, 768, 256,
      nullptr, nullptr, nullptr, qb, kbuf, vtb);

  // 4) Windowed attention (hbuf reused as attnout)
  attn_kernel<<<NWIN, 256, 0, stream>>>(qb, kbuf, vtb, hbuf);

  // 5) proj GEMM + bias + residual(x) -> d_out (x_new, fp32)
  gemm_kernel<0><<<(ROWS / 256) * (CDIM / 64), 256, 0, stream>>>(
      hbuf, wpp, proj_b, ROWS, CDIM, 256,
      x, out, nullptr, nullptr, nullptr, nullptr);

  // 6) LN2 on x_new -> bf16 (reuses Q region)
  ln_kernel<<<ROWS / 8, 256, 0, stream>>>(out, ln2_g, ln2_b, h2, ROWS);

  // 7) MLP up-proj + exact GELU -> bf16 a1
  gemm_kernel<1><<<(ROWS / 256) * (HDIM / 64), 256, 0, stream>>>(
      h2, w1p, b1, ROWS, HDIM, 256,
      nullptr, nullptr, a1, nullptr, nullptr, nullptr);

  // 8) MLP down-proj + bias + residual(x_new in d_out) -> d_out
  gemm_kernel<0><<<(ROWS / 256) * (CDIM / 64), 256, 0, stream>>>(
      a1, w2p, b2, ROWS, CDIM, 1024,
      out, out, nullptr, nullptr, nullptr, nullptr);
}
